// GAT_73950746902569
// MI455X (gfx1250) — compile-verified, matrix operand order checked by reading, not run on hardware
//
#include <hip/hip_runtime.h>
#include <hip/hip_bf16.h>

// ---------------- problem constants ----------------
#define NN   100000          // nodes
#define EEC  200000          // edges
#define BBC  4096            // graphs
#define ETOT (NN + EEC)      // edges + self loops
#define HIDC 256

// ---------------- WMMA types ----------------
typedef __attribute__((ext_vector_type(16))) __bf16 v16bf;
typedef __attribute__((ext_vector_type(8)))  float  v8f;

union FragBF16 {
    v16bf v;
    uint4 q[2];
};

// monotonic float <-> uint encoding for atomicMax on signed floats
__device__ __forceinline__ unsigned int fenc(float f) {
    unsigned int u = __float_as_uint(f);
    return (u & 0x80000000u) ? ~u : (u | 0x80000000u);
}
__device__ __forceinline__ float fdec(unsigned int u) {
    return (u & 0x80000000u) ? __uint_as_float(u ^ 0x80000000u)
                             : __uint_as_float(~u);
}

// ---------------- generic helpers ----------------
__global__ void k_zero_f32(float* p, long long n) {
    long long i = (long long)blockIdx.x * 256 + threadIdx.x;
    if (i < n) p[i] = 0.0f;
}

__global__ void k_f32_to_bf16(const float* __restrict__ in, __bf16* __restrict__ out, long long n) {
    long long i = (long long)blockIdx.x * 256 + threadIdx.x;
    if (i < n) out[i] = (__bf16)in[i];
}

// W [K, Nc] f32 row-major  ->  Wt [Nc, K] bf16 row-major (transposed for B frags)
__global__ void k_transpose_bf16(const float* __restrict__ W, __bf16* __restrict__ Wt, int K, int Nc) {
    long long idx = (long long)blockIdx.x * 256 + threadIdx.x;
    if (idx >= (long long)K * Nc) return;
    int n = (int)(idx / K);
    int k = (int)(idx % K);
    Wt[idx] = (__bf16)W[(size_t)k * Nc + n];
}

// ---------------- bf16 WMMA GEMM ----------------
// C[M,Nc] = A[M,K](bf16,row-major) * Bt[Nc,K](bf16,row-major, pre-transposed) + bias
// block = 256 threads (8 waves, 4(M) x 2(N)), block tile 128x128, wave tile 32x64.
// Wide N-tile halves A re-read amplification (A is the large streamed operand;
// Bt is the small L2-resident weight matrix).
template <int ACT, bool WB16>
__global__ __launch_bounds__(256)
void k_gemm_bf16_wmma(const __bf16* __restrict__ A, const __bf16* __restrict__ Bt,
                      const float* __restrict__ bias,
                      float* __restrict__ C, __bf16* __restrict__ Cb,
                      int M, int Nc, int K) {
    const int lane = threadIdx.x & 31;
    const int w    = threadIdx.x >> 5;
    const int wm   = w & 3;
    const int wn   = w >> 2;
    const int m0   = blockIdx.x * 128 + wm * 32;
    const int n0   = blockIdx.y * 128 + wn * 64;
    const int rA   = lane & 15;
    const int kb   = (lane >> 4) * 8;   // K sub-offset per documented 16-bit A layout

    v8f acc[2][4] = {};

    for (int k0 = 0; k0 < K; k0 += 32) {
        FragBF16 a[2], b[4];
#pragma unroll
        for (int mi = 0; mi < 2; ++mi) {
            int row = m0 + mi * 16 + rA;
            row = row < M ? row : M - 1;                    // clamp (stores are guarded)
            const __bf16* p = A + (size_t)row * K + (k0 + kb);
            a[mi].q[0] = *(const uint4*)(p);                // K = kb+0..7
            a[mi].q[1] = *(const uint4*)(p + 16);           // K = kb+16..23
        }
#pragma unroll
        for (int ni = 0; ni < 4; ++ni) {
            int col = n0 + ni * 16 + rA;                    // Nc is a multiple of 128
            const __bf16* p = Bt + (size_t)col * K + (k0 + kb);
            b[ni].q[0] = *(const uint4*)(p);
            b[ni].q[1] = *(const uint4*)(p + 16);
        }
#pragma unroll
        for (int mi = 0; mi < 2; ++mi)
#pragma unroll
            for (int ni = 0; ni < 4; ++ni)
                acc[mi][ni] = __builtin_amdgcn_wmma_f32_16x16x32_bf16(
                    false, a[mi].v, false, b[ni].v, (short)0, acc[mi][ni], false, false);
    }

    const int cr = (lane >> 4) * 8;   // D layout: M = r + (lane/16)*8
    const int cc = lane & 15;         // N = lane % 16
#pragma unroll
    for (int mi = 0; mi < 2; ++mi) {
#pragma unroll
        for (int ni = 0; ni < 4; ++ni) {
            int   col = n0 + ni * 16 + cc;
            float bv  = bias ? bias[col] : 0.0f;
#pragma unroll
            for (int r = 0; r < 8; ++r) {
                int row = m0 + mi * 16 + cr + r;
                if (row < M) {
                    float v = acc[mi][ni][r] + bv;
                    if (ACT == 1) v = fmaxf(v, 0.0f);
                    C[(size_t)row * Nc + col] = v;
                    if (WB16) Cb[(size_t)row * Nc + col] = (__bf16)v;
                }
            }
        }
    }
}

// ---------------- featurization ----------------
// xn[n] = [atom_emb[x[n,0]] (128) | unpackbits(x[n,1:8]) @ atom_lin_w + b (128)]
__global__ void k_node_feat(const int* __restrict__ x, const float* __restrict__ aemb,
                            const float* __restrict__ alw, const float* __restrict__ alb,
                            float* __restrict__ xn, __bf16* __restrict__ xnb) {
    size_t n = blockIdx.x;
    int    j = threadIdx.x;  // 128
    __shared__ unsigned long long mask;
    __shared__ int a0s;
    if (j == 0) {
        a0s = x[n * 8];
        unsigned long long m = 0;
        for (int c = 0; c < 7; ++c) {
            int v = x[n * 8 + 1 + c];
            for (int b = 0; b < 8; ++b)
                if ((v >> (7 - b)) & 1) m |= 1ull << (c * 8 + b);  // MSB-first bits
        }
        mask = m;
    }
    __syncthreads();
    float e = aemb[(size_t)a0s * 128 + j];
    float s = alb[j];
    unsigned long long m = mask;
    for (int k = 0; k < 56; ++k)
        if ((m >> k) & 1) s += alw[k * 128 + j];
    xn[n * 256 + j]        = e;
    xn[n * 256 + 128 + j]  = s;
    xnb[n * 256 + j]       = (__bf16)e;
    xnb[n * 256 + 128 + j] = (__bf16)s;
}

// ef[e] = [edge_emb[ea[e,0]] (64) | unpackbits(ea[e,1]) @ edge_lin_w + b (64)]
__global__ void k_edge_feat(const int* __restrict__ ea, const float* __restrict__ eemb,
                            const float* __restrict__ elw, const float* __restrict__ elb,
                            float* __restrict__ ef) {
    size_t e = blockIdx.x;
    int    t = threadIdx.x;  // 128
    int a0 = ea[e * 2], a1 = ea[e * 2 + 1];
    if (t < 64) {
        ef[e * 128 + t] = eemb[(size_t)a0 * 64 + t];
    } else {
        int   j = t - 64;
        float s = elb[j];
        for (int k = 0; k < 8; ++k)
            if ((a1 >> (7 - k)) & 1) s += elw[k * 64 + j];
        ef[e * 128 + 64 + j] = s;
    }
}

// self-loop attr = mean of incoming edge attrs (accumulate into ef rows [EEC, EEC+NN))
__global__ void k_loop_acc(float* __restrict__ ef, const int* __restrict__ tgtE,
                           float* __restrict__ cnt) {
    size_t e  = blockIdx.x;
    int    t  = threadIdx.x;  // 128
    int    tg = tgtE[e];
    atomicAdd(&ef[((size_t)EEC + tg) * 128 + t], ef[e * 128 + t]);
    if (t == 0) atomicAdd(&cnt[tg], 1.0f);
}

__global__ void k_loop_div(float* __restrict__ ef, const float* __restrict__ cnt) {
    size_t n = blockIdx.x;
    int    t = threadIdx.x;  // 128
    ef[((size_t)EEC + n) * 128 + t] *= 1.0f / fmaxf(cnt[n], 1.0f);
}

// ---------------- GATv2 attention ----------------
// alpha[i,h] = sum_c att[h,c] * leaky_relu(xl[src,h,c] + xr[tgt,h,c] + ee[i,h,c], 0.2)
__global__ void k_att_score(const float* __restrict__ xl, const float* __restrict__ xr,
                            const float* __restrict__ ee,
                            const int* __restrict__ srcE, const int* __restrict__ tgtE,
                            const float* __restrict__ attv,
                            float* __restrict__ alpha, unsigned int* __restrict__ amax,
                            int Etot, int Ee, int H) {
    int       wid  = threadIdx.x >> 5;
    int       lane = threadIdx.x & 31;
    long long i    = (long long)blockIdx.x * 8 + wid;
    if (i >= Etot) return;
    int s = (i < Ee) ? srcE[i] : (int)(i - Ee);
    int t = (i < Ee) ? tgtE[i] : (int)(i - Ee);
    const float* pl = xl + (size_t)s * 256;
    const float* pr = xr + (size_t)t * 256;
    const float* pe = ee + (size_t)i * 256;
    float v[8];
#pragma unroll
    for (int k = 0; k < 8; ++k) {
        int   c = k * 32 + lane;
        float m = pl[c] + pr[c] + pe[c];
        m       = (m > 0.0f) ? m : 0.2f * m;
        v[k]    = attv[c] * m;
    }
    if (H == 8) {
#pragma unroll
        for (int k = 0; k < 8; ++k) {   // channels k*32+lane all belong to head k
            float r = v[k];
            for (int o = 16; o > 0; o >>= 1) r += __shfl_xor(r, o, 32);
            if (lane == 0) {
                alpha[i * 8 + k] = r;
                atomicMax(&amax[(size_t)t * 8 + k], fenc(r));
            }
        }
    } else {
        float r = 0.0f;
#pragma unroll
        for (int k = 0; k < 8; ++k) r += v[k];
        for (int o = 16; o > 0; o >>= 1) r += __shfl_xor(r, o, 32);
        if (lane == 0) {
            alpha[i] = r;
            atomicMax(&amax[t], fenc(r));
        }
    }
}

__global__ void k_att_exp(float* __restrict__ alpha, const unsigned int* __restrict__ amax,
                          const int* __restrict__ tgtE, float* __restrict__ asum,
                          long long total, int Ee, int H) {
    long long idx = (long long)blockIdx.x * 256 + threadIdx.x;
    if (idx >= total) return;
    long long i = idx / H;
    int       h = (int)(idx % H);
    int       t = (i < Ee) ? tgtE[i] : (int)(i - Ee);
    float     p = expf(alpha[idx] - fdec(amax[(size_t)t * H + h]));
    alpha[idx]  = p;
    atomicAdd(&asum[(size_t)t * H + h], p);
}

__global__ void k_att_agg(const float* __restrict__ xl, const float* __restrict__ alpha,
                          const float* __restrict__ asum,
                          const int* __restrict__ srcE, const int* __restrict__ tgtE,
                          float* __restrict__ outb, int Ee, int H) {
    size_t i = blockIdx.x;
    int    c = threadIdx.x;  // 256
    int s = (i < (size_t)Ee) ? srcE[i] : (int)(i - Ee);
    int t = (i < (size_t)Ee) ? tgtE[i] : (int)(i - Ee);
    int h = (H == 8) ? (c >> 5) : 0;
    float w = alpha[i * H + h] / asum[(size_t)t * H + h];
    atomicAdd(&outb[(size_t)t * 256 + c], xl[(size_t)s * 256 + c] * w);
}

// ---------------- BatchNorm (training-mode stats; conv_b cancels) ----------------
__global__ void k_bn_stats(const float* __restrict__ outb, float* __restrict__ sums, int Nrows) {
    int       c    = threadIdx.x;  // 256
    long long base = (long long)blockIdx.x * 256;
    float s = 0.0f, s2 = 0.0f;
    for (int r = 0; r < 256; ++r) {
        long long row = base + r;
        if (row >= Nrows) break;
        float v = outb[row * 256 + c];
        s += v;
        s2 += v * v;
    }
    atomicAdd(&sums[c], s);
    atomicAdd(&sums[256 + c], s2);
}

__global__ void k_bn_final(const float* __restrict__ sums, const float* __restrict__ g,
                           const float* __restrict__ b, float* __restrict__ sc,
                           float* __restrict__ sh, float invN) {
    int   c   = threadIdx.x;  // 256
    float mu  = sums[c] * invN;
    float var = sums[256 + c] * invN - mu * mu;
    float s   = g[c] * rsqrtf(var + 1e-5f);
    sc[c] = s;
    sh[c] = b[c] - mu * s;
}

__global__ void k_bn_norm(const float* __restrict__ outb, const float* __restrict__ sc,
                          const float* __restrict__ sh, float* __restrict__ xn,
                          __bf16* __restrict__ xnb, int last) {
    size_t n = blockIdx.x;
    int    c = threadIdx.x;  // 256
    float  y = outb[n * 256 + c] * sc[c] + sh[c];
    if (!last) y = (y > 0.0f) ? y : 0.01f * y;
    xn[n * 256 + c]  = y;
    xnb[n * 256 + c] = (__bf16)y;
}

// ---------------- global mean pool + head ----------------
__global__ void k_pool_acc(const float* __restrict__ xn, const int* __restrict__ batch,
                           float* __restrict__ g, float* __restrict__ gcnt) {
    size_t n = blockIdx.x;
    int    c = threadIdx.x;  // 256
    int    b = batch[n];
    atomicAdd(&g[(size_t)b * 256 + c], xn[n * 256 + c]);
    if (c == 0) atomicAdd(&gcnt[b], 1.0f);
}

__global__ void k_pool_div(float* __restrict__ g, const float* __restrict__ gcnt,
                           __bf16* __restrict__ gb) {
    size_t b = blockIdx.x;
    int    c = threadIdx.x;  // 256
    float  v = g[b * 256 + c] / fmaxf(gcnt[b], 1.0f);
    g[b * 256 + c]  = v;
    gb[b * 256 + c] = (__bf16)v;
}

__global__ void k_mlp_final(const float* __restrict__ h3, const float* __restrict__ w4,
                            const float* __restrict__ b4, float* __restrict__ out) {
    int b    = blockIdx.x;
    int j    = threadIdx.x >> 5;  // block = 96 threads = 3 waves
    int lane = threadIdx.x & 31;
    float s = 0.0f;
    for (int k = lane; k < 512; k += 32) s += h3[(size_t)b * 512 + k] * w4[k * 3 + j];
    for (int o = 16; o > 0; o >>= 1) s += __shfl_xor(s, o, 32);
    if (lane == 0 && j < 3) out[b * 3 + j] = s + b4[j];
}

// ---------------- host orchestration ----------------
#define ZERO(ptr, n)                                                                     \
    k_zero_f32<<<dim3((unsigned)((((long long)(n)) + 255) / 256)), 256, 0, stream>>>(    \
        (float*)(ptr), (long long)(n))

extern "C" void kernel_launch(void* const* d_in, const int* in_sizes, int n_in,
                              void* d_out, int out_size, void* d_ws, size_t ws_size,
                              hipStream_t stream) {
    // -------- inputs (setup_inputs order) --------
    const int*   x     = (const int*)d_in[0];
    const int*   eidx  = (const int*)d_in[1];
    const int*   eattr = (const int*)d_in[2];
    const int*   batch = (const int*)d_in[3];
    const float* aemb  = (const float*)d_in[4];
    const float* alw   = (const float*)d_in[5];
    const float* albv  = (const float*)d_in[6];
    const float* eemb  = (const float*)d_in[7];
    const float* elw   = (const float*)d_in[8];
    const float* elbv  = (const float*)d_in[9];
    const float* linlw = (const float*)d_in[10];
    const float* linlb = (const float*)d_in[11];
    const float* linrw = (const float*)d_in[12];
    const float* linrb = (const float*)d_in[13];
    const float* linew = (const float*)d_in[14];
    const float* attw  = (const float*)d_in[15];
    // d_in[16] = conv_b : cancels exactly inside training-mode BatchNorm
    const float* bng = (const float*)d_in[17];
    const float* bnb = (const float*)d_in[18];
    const float* w1  = (const float*)d_in[19];
    const float* b1  = (const float*)d_in[20];
    const float* w2  = (const float*)d_in[21];
    const float* b2  = (const float*)d_in[22];
    const float* w3  = (const float*)d_in[23];
    const float* b3  = (const float*)d_in[24];
    const float* w4  = (const float*)d_in[25];
    const float* b4  = (const float*)d_in[26];
    float*       out = (float*)d_out;

    const int* srcE = eidx;
    const int* tgtE = eidx + EEC;

    // -------- workspace bump allocator --------
    char*  ws  = (char*)d_ws;
    size_t off = 0;
    auto   alloc = [&](size_t bytes) -> void* {
        void* p = ws + off;
        off     = (off + bytes + 255) & ~(size_t)255;
        return p;
    };
    float*   xn    = (float*)alloc((size_t)NN * 256 * 4);
    __bf16*  xnb   = (__bf16*)alloc((size_t)NN * 256 * 2);
    float*   ef    = (float*)alloc((size_t)ETOT * 128 * 4);
    __bf16*  efb   = (__bf16*)alloc((size_t)ETOT * 128 * 2);
    float*   cnt   = (float*)alloc((size_t)NN * 4);
    float*   xl    = (float*)alloc((size_t)NN * 256 * 4);
    float*   xr    = (float*)alloc((size_t)NN * 256 * 4);
    float*   ee    = (float*)alloc((size_t)ETOT * 256 * 4);
    float*   alpha = (float*)alloc((size_t)ETOT * 8 * 4);
    float*   amax  = (float*)alloc((size_t)NN * 8 * 4);
    float*   asum  = (float*)alloc((size_t)NN * 8 * 4);
    float*   outb  = (float*)alloc((size_t)NN * 256 * 4);
    float*   bnsums  = (float*)alloc(512 * 4);
    float*   bnscale = (float*)alloc(256 * 4);
    float*   bnshift = (float*)alloc(256 * 4);
    float*   g     = (float*)alloc((size_t)BBC * 256 * 4);
    __bf16*  gb    = (__bf16*)alloc((size_t)BBC * 256 * 2);
    float*   gcnt  = (float*)alloc((size_t)BBC * 4);
    float*   h1    = (float*)alloc((size_t)BBC * 1024 * 4);
    __bf16*  h1b   = (__bf16*)alloc((size_t)BBC * 1024 * 2);
    float*   h2    = (float*)alloc((size_t)BBC * 1024 * 4);
    __bf16*  h2b   = (__bf16*)alloc((size_t)BBC * 1024 * 2);
    float*   h3    = (float*)alloc((size_t)BBC * 512 * 4);
    __bf16*  wt_l  = (__bf16*)alloc((size_t)4 * 256 * 256 * 2);
    __bf16*  wt_r  = (__bf16*)alloc((size_t)4 * 256 * 256 * 2);
    __bf16*  wt_e  = (__bf16*)alloc((size_t)4 * 256 * 128 * 2);
    __bf16*  w1t   = (__bf16*)alloc((size_t)1024 * 256 * 2);
    __bf16*  w2t   = (__bf16*)alloc((size_t)1024 * 1024 * 2);
    __bf16*  w3t   = (__bf16*)alloc((size_t)512 * 1024 * 2);
    (void)ws_size; (void)in_sizes; (void)n_in; (void)out_size;

    // -------- weight prep (f32 -> transposed bf16) --------
    for (int l = 0; l < 4; ++l) {
        k_transpose_bf16<<<(65536 + 255) / 256, 256, 0, stream>>>(linlw + l * 65536, wt_l + l * 65536, 256, 256);
        k_transpose_bf16<<<(65536 + 255) / 256, 256, 0, stream>>>(linrw + l * 65536, wt_r + l * 65536, 256, 256);
        k_transpose_bf16<<<(32768 + 255) / 256, 256, 0, stream>>>(linew + l * 32768, wt_e + l * 32768, 128, 256);
    }
    k_transpose_bf16<<<(262144 + 255) / 256, 256, 0, stream>>>(w1, w1t, 256, 1024);
    k_transpose_bf16<<<(1048576 + 255) / 256, 256, 0, stream>>>(w2, w2t, 1024, 1024);
    k_transpose_bf16<<<(524288 + 255) / 256, 256, 0, stream>>>(w3, w3t, 1024, 512);

    // -------- featurization + self-loop attrs --------
    ZERO(ef + (size_t)EEC * 128, (size_t)NN * 128);
    ZERO(cnt, NN);
    k_node_feat<<<NN, 128, 0, stream>>>(x, aemb, alw, albv, xn, xnb);
    k_edge_feat<<<EEC, 128, 0, stream>>>(eattr, eemb, elw, elbv, ef);
    k_loop_acc<<<EEC, 128, 0, stream>>>(ef, tgtE, cnt);
    k_loop_div<<<NN, 128, 0, stream>>>(ef, cnt);
    k_f32_to_bf16<<<(unsigned)(((long long)ETOT * 128 + 255) / 256), 256, 0, stream>>>(
        ef, efb, (long long)ETOT * 128);

    // -------- GATv2 layers --------
    dim3 gN((NN + 127) / 128, 256 / 128);    // 100k x 256
    dim3 gE((ETOT + 127) / 128, 256 / 128);  // 300k x 256
    for (int l = 0; l < 4; ++l) {
        int H = (l == 0) ? 8 : 1;
        k_gemm_bf16_wmma<0, false><<<gN, 256, 0, stream>>>(xnb, wt_l + l * 65536, linlb + l * 256,
                                                           xl, nullptr, NN, 256, 256);
        k_gemm_bf16_wmma<0, false><<<gN, 256, 0, stream>>>(xnb, wt_r + l * 65536, linrb + l * 256,
                                                           xr, nullptr, NN, 256, 256);
        k_gemm_bf16_wmma<0, false><<<gE, 256, 0, stream>>>(efb, wt_e + l * 32768, nullptr,
                                                           ee, nullptr, ETOT, 256, 128);

        ZERO(amax, (size_t)NN * 8);   // 0 == encoded -inf sentinel
        ZERO(asum, (size_t)NN * 8);
        ZERO(outb, (size_t)NN * 256);

        k_att_score<<<(ETOT + 7) / 8, 256, 0, stream>>>(xl, xr, ee, srcE, tgtE, attw + l * 256,
                                                        alpha, (unsigned int*)amax, ETOT, EEC, H);
        long long tot = (long long)ETOT * H;
        k_att_exp<<<(unsigned)((tot + 255) / 256), 256, 0, stream>>>(
            alpha, (const unsigned int*)amax, tgtE, asum, tot, EEC, H);
        k_att_agg<<<ETOT, 256, 0, stream>>>(xl, alpha, asum, srcE, tgtE, outb, EEC, H);

        ZERO(bnsums, 512);
        k_bn_stats<<<(NN + 255) / 256, 256, 0, stream>>>(outb, bnsums, NN);
        k_bn_final<<<1, 256, 0, stream>>>(bnsums, bng + l * 256, bnb + l * 256,
                                          bnscale, bnshift, 1.0f / NN);
        k_bn_norm<<<NN, 256, 0, stream>>>(outb, bnscale, bnshift, xn, xnb, l == 3);
    }

    // -------- global mean pool --------
    ZERO(g, (size_t)BBC * 256);
    ZERO(gcnt, BBC);
    k_pool_acc<<<NN, 256, 0, stream>>>(xn, batch, g, gcnt);
    k_pool_div<<<BBC, 256, 0, stream>>>(g, gcnt, gb);

    // -------- MLP head --------
    dim3 g1(BBC / 128, 1024 / 128);
    dim3 g3(BBC / 128, 512 / 128);
    k_gemm_bf16_wmma<1, true><<<g1, 256, 0, stream>>>(gb, w1t, b1, h1, h1b, BBC, 1024, 256);
    k_gemm_bf16_wmma<1, true><<<g1, 256, 0, stream>>>(h1b, w2t, b2, h2, h2b, BBC, 1024, 1024);
    k_gemm_bf16_wmma<1, false><<<g3, 256, 0, stream>>>(h2b, w3t, b3, h3, nullptr, BBC, 512, 1024);
    k_mlp_final<<<BBC, 96, 0, stream>>>(h3, w4, b4, out);
}
